// cross_attn_fusion__26757646254171
// MI455X (gfx1250) — matrix operand order repaired and checked by reading, not compiled, not run on hardware
//
#include <hip/hip_runtime.h>
#include <hip/hip_bf16.h>
#include <math.h>

#define DIM    512
#define HDIM   128
#define NHEADS 4
#define BH     32
#define NQ     512
#define NKV    1024
#define QK_SCALE 0.08838834764831845f   /* 1/sqrt(128) */

typedef __bf16 bf16_t;
typedef __attribute__((ext_vector_type(16))) __bf16 v16bf;
typedef __attribute__((ext_vector_type(8)))  __bf16 v8bf;
typedef __attribute__((ext_vector_type(8)))  float  v8f;
typedef __attribute__((ext_vector_type(4)))  float  v4f;

// ---------------------------------------------------------------------------
// WMMA helpers: split-bf16 emulated-fp32 matmul (3x bf16 WMMA per K=32 step)
// ---------------------------------------------------------------------------
static __device__ __forceinline__ v8f wmma_bf(v16bf a, v16bf b, v8f c) {
  return __builtin_amdgcn_wmma_f32_16x16x32_bf16(false, a, false, b,
                                                 (short)0, c, false, false);
}
static __device__ __forceinline__ v8f wmma3(v16bf ah, v16bf al,
                                            v16bf bh, v16bf bl, v8f c) {
  c = wmma_bf(ah, bh, c);
  c = wmma_bf(al, bh, c);
  c = wmma_bf(ah, bl, c);
  return c;
}
static __device__ __forceinline__ void split1(float x, bf16_t& h, bf16_t& l) {
  h = (bf16_t)x;
  l = (bf16_t)(x - (float)h);
}

// A fragment from a pre-split bf16 row: elems 0..15 <- K=kb..kb+15,
// kb=(lane>>4)*16 (contiguous order, matches B).
static __device__ __forceinline__ v16bf loadA_bf(const bf16_t* __restrict__ rowp,
                                                 int lane) {
  const bf16_t* p = rowp + ((lane >> 4) * 16);
  v8bf u0 = *(const v8bf*)p;
  v8bf u1 = *(const v8bf*)(p + 8);
  return __builtin_shufflevector(u0, u1, 0, 1, 2, 3, 4, 5, 6, 7,
                                 8, 9, 10, 11, 12, 13, 14, 15);
}
// B fragment (NT): 16 contiguous bf16 along K -> single 32B load.
static __device__ __forceinline__ v16bf loadB_bf(const bf16_t* __restrict__ p) {
  return *(const v16bf*)p;
}

// ---------------------------------------------------------------------------
// Kernel 0: one-shot f32 -> (hi,lo) bf16 split (for weights). n % 8 == 0.
// ---------------------------------------------------------------------------
__global__ __launch_bounds__(256) void split_f32_kernel(
    const float* __restrict__ X, bf16_t* __restrict__ H,
    bf16_t* __restrict__ L, int n) {
  int i = (blockIdx.x * 256 + threadIdx.x) * 8;
  if (i >= n) return;
  v4f x0 = *(const v4f*)(X + i);
  v4f x1 = *(const v4f*)(X + i + 4);
  v8bf h, l;
#pragma unroll
  for (int j = 0; j < 4; ++j) {
    bf16_t hh, ll;
    split1(x0[j], hh, ll); h[j] = hh; l[j] = ll;
    split1(x1[j], hh, ll); h[4 + j] = hh; l[4 + j] = ll;
  }
  *(v8bf*)(H + i) = h;
  *(v8bf*)(L + i) = l;
}

// ---------------------------------------------------------------------------
// Kernel 1: fused LayerNorm + pos_emb (pos batch = i%4 quirk) + modality emb,
// writing (hi,lo) bf16. One wave32 per 512-wide row; 8 rows per block.
// ---------------------------------------------------------------------------
__global__ __launch_bounds__(256) void ln_pos_me_kernel(
    const float* __restrict__ X, const float* __restrict__ pos,
    const float* __restrict__ gamma, const float* __restrict__ beta,
    const float* __restrict__ me, bf16_t* __restrict__ Yh,
    bf16_t* __restrict__ Yl) {
  int lane = threadIdx.x & 31;
  int w = threadIdx.x >> 5;
  int rowg = blockIdx.x * 8 + w;      // 0..BH*NQ-1
  int batch = rowg >> 9;              // /512
  int row = rowg & 511;

  const float* x = X + (size_t)rowg * DIM + lane * 16;
  v4f v[4];
  float s = 0.f, ss = 0.f;
#pragma unroll
  for (int q = 0; q < 4; ++q) {
    v[q] = *(const v4f*)(x + 4 * q);
#pragma unroll
    for (int j = 0; j < 4; ++j) { s += v[q][j]; ss += v[q][j] * v[q][j]; }
  }
#pragma unroll
  for (int m = 1; m < 32; m <<= 1) {
    s += __shfl_xor(s, m);
    ss += __shfl_xor(ss, m);
  }
  float mean = s * (1.0f / DIM);
  float var = ss * (1.0f / DIM) - mean * mean;
  float rstd = rsqrtf(var + 1e-5f);

  const float* pp = pos + ((size_t)(batch & 3) * NQ + row) * DIM + lane * 16;
  v8bf oh[2], ol[2];
#pragma unroll
  for (int q = 0; q < 4; ++q) {
    v4f pv = *(const v4f*)(pp + 4 * q);
#pragma unroll
    for (int j = 0; j < 4; ++j) {
      int c = lane * 16 + 4 * q + j;
      float y = (v[q][j] - mean) * rstd * gamma[c] + beta[c] + pv[j] + me[c];
      bf16_t hh, ll; split1(y, hh, ll);
      oh[q >> 1][(q & 1) * 4 + j] = hh;
      ol[q >> 1][(q & 1) * 4 + j] = ll;
    }
  }
  size_t off = (size_t)rowg * DIM + lane * 16;
  *(v8bf*)(Yh + off) = oh[0];
  *(v8bf*)(Yh + off + 8) = oh[1];
  *(v8bf*)(Yl + off) = ol[0];
  *(v8bf*)(Yl + off + 8) = ol[1];
}

// ---------------------------------------------------------------------------
// Kernel 2/4: C[M,512] = A[M,512] @ W[512,512]^T + bias with pre-split bf16
// operands. A1 pair != nullptr => per-batch concat rows [A0_b(512);A1_b(512)].
// WRITE_F32: straight f32 output; else (hi,lo) bf16 pair (branch-free stores).
// Block 256 thr = 8 waves (4 along M x 2 along N); block tile 128x128;
// wave tile 32x64 (2x4 WMMA subtiles) -> 24 WMMAs per 12 fragment loads.
// ---------------------------------------------------------------------------
template <bool WRITE_F32>
__global__ __launch_bounds__(256) void gemm_nt_bias_kernel(
    const bf16_t* __restrict__ Ah0, const bf16_t* __restrict__ Al0,
    const bf16_t* __restrict__ Ah1, const bf16_t* __restrict__ Al1,
    const bf16_t* __restrict__ Wh, const bf16_t* __restrict__ Wl,
    const float* __restrict__ bias, float* __restrict__ Cf32,
    bf16_t* __restrict__ Chi, bf16_t* __restrict__ Clo) {
  int lane = threadIdx.x & 31;
  int w = threadIdx.x >> 5;
  int mbase = blockIdx.x * 128 + (w & 3) * 32;   // 4 waves along M
  int nbase = blockIdx.y * 128 + (w >> 2) * 64;  // 2 waves along N

  const bf16_t* arow_h[2];
  const bf16_t* arow_l[2];
#pragma unroll
  for (int i = 0; i < 2; ++i) {
    int r = mbase + i * 16 + (lane & 15);
    if (Ah1) {
      int batch = r >> 10;            // 1024 concat rows per batch
      int sub = r & 1023;
      size_t off = (sub < NQ) ? ((size_t)(batch * NQ + sub)) * DIM
                              : ((size_t)(batch * NQ + sub - NQ)) * DIM;
      arow_h[i] = (sub < NQ ? Ah0 : Ah1) + off;
      arow_l[i] = (sub < NQ ? Al0 : Al1) + off;
    } else {
      arow_h[i] = Ah0 + (size_t)r * DIM;
      arow_l[i] = Al0 + (size_t)r * DIM;
    }
  }
  const bf16_t* brow_h[4];
  const bf16_t* brow_l[4];
#pragma unroll
  for (int j = 0; j < 4; ++j) {
    size_t off = (size_t)(nbase + j * 16 + (lane & 15)) * DIM + (lane >> 4) * 16;
    brow_h[j] = Wh + off;
    brow_l[j] = Wl + off;
  }

  v8f acc[2][4] = {};
  for (int k = 0; k < DIM; k += 32) {
    // prefetch next K slab of the streamed operands
    if (k + 32 < DIM) {
      __builtin_prefetch(arow_h[0] + k + 32, 0, 3);
      __builtin_prefetch(brow_h[0] + k + 32, 0, 3);
    }
    v16bf ah[2], al[2], bh[4], bl[4];
#pragma unroll
    for (int i = 0; i < 2; ++i) {
      ah[i] = loadA_bf(arow_h[i] + k, lane);
      al[i] = loadA_bf(arow_l[i] + k, lane);
    }
#pragma unroll
    for (int j = 0; j < 4; ++j) {
      bh[j] = loadB_bf(brow_h[j] + k);
      bl[j] = loadB_bf(brow_l[j] + k);
    }
#pragma unroll
    for (int i = 0; i < 2; ++i)
#pragma unroll
      for (int j = 0; j < 4; ++j)
        acc[i][j] = wmma3(ah[i], al[i], bh[j], bl[j], acc[i][j]);
  }

  int hl = lane >> 4;
#pragma unroll
  for (int i = 0; i < 2; ++i)
#pragma unroll
    for (int j = 0; j < 4; ++j) {
      int col = nbase + j * 16 + (lane & 15);
      float bv = bias[col];
#pragma unroll
      for (int r = 0; r < 8; ++r) {
        int row = mbase + i * 16 + 8 * hl + r;
        float val = acc[i][j][r] + bv;
        size_t idx = (size_t)row * DIM + col;
        if constexpr (WRITE_F32) {
          Cf32[idx] = val;
        } else {
          bf16_t hh, ll; split1(val, hh, ll);
          Chi[idx] = hh; Clo[idx] = ll;
        }
      }
    }
}

// ---------------------------------------------------------------------------
// Kernel 3: flash-style attention per (batch b, head h), pre-split bf16 Q/K.
// Block 256 thr = 8 waves; each wave owns 16 query rows and a full 16x128
// f32 O accumulator; loop over 32-key tiles with online softmax.
// V (f32 in) staged transposed (hi/lo bf16) in LDS once per block per tile;
// P staged per-wave in LDS for C-layout -> A-layout (same-wave DS ordering).
// Output O written as (hi,lo) bf16 for the Wo GEMM.
// ---------------------------------------------------------------------------
__global__ __launch_bounds__(256) void attn_kernel(
    const bf16_t* __restrict__ Qh, const bf16_t* __restrict__ Ql,   // [BH,NQ,DIM]
    const bf16_t* __restrict__ Kh, const bf16_t* __restrict__ Kl,   // [BH,NKV,DIM]
    const float* __restrict__ V0,   // first 512 keys'  values (f32)
    const float* __restrict__ V1,   // second 512 keys' values (f32)
    bf16_t* __restrict__ Ohi, bf16_t* __restrict__ Olo) {
  __shared__ __align__(16) bf16_t vhiS[HDIM * 32];   // [d][key]
  __shared__ __align__(16) bf16_t vloS[HDIM * 32];
  __shared__ __align__(16) bf16_t pS[8][16 * 32];    // per-wave P staging

  int tid = threadIdx.x;
  int lane = tid & 31, w = tid >> 5;
  int b = blockIdx.z, h = blockIdx.y;
  int qrow0 = blockIdx.x * 128 + w * 16;

  // Resident Q fragments: 16 rows x 128 dims = 4 K-steps of 32.
  v16bf qfh[4], qfl[4];
  {
    size_t qo = ((size_t)b * NQ + qrow0 + (lane & 15)) * DIM + h * HDIM;
#pragma unroll
    for (int ks = 0; ks < 4; ++ks) {
      qfh[ks] = loadA_bf(Qh + qo + ks * 32, lane);
      qfl[ks] = loadA_bf(Ql + qo + ks * 32, lane);
    }
  }

  v8f o[8] = {};
  float mrow[8], lrow[8];
#pragma unroll
  for (int r = 0; r < 8; ++r) { mrow[r] = -3.0e38f; lrow[r] = 0.f; }

  for (int kt = 0; kt < NKV / 32; ++kt) {
    // ---- stage V tile (32 keys x 128 dims), transposed, hi/lo bf16 ----
#pragma unroll
    for (int it = 0; it < 4; ++it) {
      int c = it * 256 + tid;         // 1024 float4 chunks
      int krow = c >> 5;              // key within tile
      int d0 = (c & 31) * 4;          // head-dim offset
      int gkey = kt * 32 + krow;
      const float* vp =
          (gkey < NQ) ? V0 + ((size_t)b * NQ + gkey) * DIM + h * HDIM + d0
                      : V1 + ((size_t)b * NQ + (gkey - NQ)) * DIM + h * HDIM + d0;
      v4f vv = *(const v4f*)vp;
#pragma unroll
      for (int j = 0; j < 4; ++j) {
        bf16_t hh, ll; split1(vv[j], hh, ll);
        vhiS[(d0 + j) * 32 + krow] = hh;
        vloS[(d0 + j) * 32 + krow] = ll;
      }
    }
    __syncthreads();

    // ---- S = Q @ K^T for two 16x16 score tiles (32 keys) ----
    v8f s0 = {}, s1 = {};
    {
      size_t ko = ((size_t)b * NKV + kt * 32 + (lane & 15)) * DIM + h * HDIM +
                  (lane >> 4) * 16;
      if (kt + 1 < NKV / 32) {
        __builtin_prefetch(Kh + ko + 32 * DIM, 0, 3);   // next key tile
        __builtin_prefetch(Kl + ko + 32 * DIM, 0, 3);
      }
#pragma unroll
      for (int ks = 0; ks < 4; ++ks) {
        v16bf b0h = loadB_bf(Kh + ko + ks * 32);
        v16bf b0l = loadB_bf(Kl + ko + ks * 32);
        v16bf b1h = loadB_bf(Kh + ko + 16 * DIM + ks * 32);
        v16bf b1l = loadB_bf(Kl + ko + 16 * DIM + ks * 32);
        s0 = wmma3(qfh[ks], qfl[ks], b0h, b0l, s0);
        s1 = wmma3(qfh[ks], qfl[ks], b1h, b1l, s1);
      }
    }

    // ---- online softmax update ----
    int hl = lane >> 4;
#pragma unroll
    for (int r = 0; r < 8; ++r) {
      float x0 = s0[r] * QK_SCALE;
      float x1 = s1[r] * QK_SCALE;
      float mx = fmaxf(x0, x1);
#pragma unroll
      for (int msk = 1; msk < 16; msk <<= 1) mx = fmaxf(mx, __shfl_xor(mx, msk));
      float mnew = fmaxf(mrow[r], mx);
      float alpha = __expf(mrow[r] - mnew);
      mrow[r] = mnew;
      float p0 = __expf(x0 - mnew);
      float p1 = __expf(x1 - mnew);
      float ps = p0 + p1;
#pragma unroll
      for (int msk = 1; msk < 16; msk <<= 1) ps += __shfl_xor(ps, msk);
      lrow[r] = lrow[r] * alpha + ps;
#pragma unroll
      for (int t = 0; t < 8; ++t) o[t][r] *= alpha;
      int m = r + 8 * hl;
      pS[w][m * 32 + (lane & 15)] = (bf16_t)p0;
      pS[w][m * 32 + (lane & 15) + 16] = (bf16_t)p1;
    }

    // ---- read P back in A layout (same-wave DS ops are in order) ----
    v16bf pa;
    {
      const bf16_t* pp = &pS[w][(lane & 15) * 32 + (lane >> 4) * 16];
      v8bf u0 = *(const v8bf*)pp;
      v8bf u1 = *(const v8bf*)(pp + 8);
      pa = __builtin_shufflevector(u0, u1, 0, 1, 2, 3, 4, 5, 6, 7,
                                   8, 9, 10, 11, 12, 13, 14, 15);
    }

    // ---- O += P @ V (bf16 P x split-bf16 V) ----
#pragma unroll
    for (int nt = 0; nt < 8; ++nt) {
      int d = nt * 16 + (lane & 15);
      int ks2 = (lane >> 4) * 16;
      v16bf bh = *(const v16bf*)(&vhiS[d * 32 + ks2]);
      v16bf bl = *(const v16bf*)(&vloS[d * 32 + ks2]);
      o[nt] = wmma_bf(pa, bh, o[nt]);
      o[nt] = wmma_bf(pa, bl, o[nt]);
    }
    __syncthreads();
  }

  // ---- epilogue: O / l, written as (hi,lo) bf16 for the Wo GEMM ----
  int hl = lane >> 4;
#pragma unroll
  for (int nt = 0; nt < 8; ++nt) {
    int col = h * HDIM + nt * 16 + (lane & 15);
#pragma unroll
    for (int r = 0; r < 8; ++r) {
      int row = qrow0 + r + 8 * hl;
      float val = o[nt][r] / lrow[r];
      bf16_t hh, ll; split1(val, hh, ll);
      size_t idx = ((size_t)b * NQ + row) * DIM + col;
      Ohi[idx] = hh;
      Olo[idx] = ll;
    }
  }
}

// ---------------------------------------------------------------------------
// Host-side orchestration
// ---------------------------------------------------------------------------
extern "C" void kernel_launch(void* const* d_in, const int* in_sizes, int n_in,
                              void* d_out, int out_size, void* d_ws,
                              size_t ws_size, hipStream_t stream) {
  (void)in_sizes; (void)n_in; (void)out_size; (void)ws_size;

  const float* attn_rgb = (const float*)d_in[0];
  const float* attn_tir = (const float*)d_in[1];
  const float* pos_emb  = (const float*)d_in[2];
  const float* ln_g = (const float*)d_in[3];
  const float* ln_b = (const float*)d_in[4];
  const float* Wq = (const float*)d_in[5];
  const float* bq = (const float*)d_in[6];
  const float* Wk = (const float*)d_in[7];
  const float* bk = (const float*)d_in[8];
  const float* Wo = (const float*)d_in[9];
  const float* bo = (const float*)d_in[10];
  const float* me_rgb = (const float*)d_in[11];
  const float* me_tir = (const float*)d_in[12];

  float* out = (float*)d_out;
  const size_t HALF = (size_t)BH * NQ * DIM;   // 8,388,608 elements
  const size_t WSZ = (size_t)DIM * DIM;        // 262,144 elements

  bf16_t* p = (bf16_t*)d_ws;
  bf16_t* rgbn_h = p; p += HALF;
  bf16_t* rgbn_l = p; p += HALF;
  bf16_t* tirn_h = p; p += HALF;
  bf16_t* tirn_l = p; p += HALF;
  bf16_t* Kp_h   = p; p += 2 * HALF;           // [BH, NKV, DIM]
  bf16_t* Kp_l   = p; p += 2 * HALF;
  bf16_t* Q_h    = p; p += HALF;               // reused for both passes
  bf16_t* Q_l    = p; p += HALF;
  bf16_t* O_h    = p; p += HALF;
  bf16_t* O_l    = p; p += HALF;
  bf16_t* Wq_h   = p; p += WSZ;
  bf16_t* Wq_l   = p; p += WSZ;
  bf16_t* Wk_h   = p; p += WSZ;
  bf16_t* Wk_l   = p; p += WSZ;
  bf16_t* Wo_h   = p; p += WSZ;
  bf16_t* Wo_l   = p; p += WSZ;

  dim3 blk(256);

  // 0) one-shot weight splits
  split_f32_kernel<<<dim3(WSZ / (256 * 8)), blk, 0, stream>>>(Wq, Wq_h, Wq_l, (int)WSZ);
  split_f32_kernel<<<dim3(WSZ / (256 * 8)), blk, 0, stream>>>(Wk, Wk_h, Wk_l, (int)WSZ);
  split_f32_kernel<<<dim3(WSZ / (256 * 8)), blk, 0, stream>>>(Wo, Wo_h, Wo_l, (int)WSZ);

  // 1) LayerNorm + pos + modality embedding -> (hi,lo) bf16
  ln_pos_me_kernel<<<dim3(BH * NQ / 8), blk, 0, stream>>>(
      attn_rgb, pos_emb, ln_g, ln_b, me_rgb, rgbn_h, rgbn_l);
  ln_pos_me_kernel<<<dim3(BH * NQ / 8), blk, 0, stream>>>(
      attn_tir, pos_emb, ln_g, ln_b, me_tir, tirn_h, tirn_l);

  // 2) K projection over per-batch concat(rgb_n, tir_n) -> Kp (hi,lo)
  gemm_nt_bias_kernel<false><<<dim3(BH * NKV / 128, DIM / 128), blk, 0, stream>>>(
      rgbn_h, rgbn_l, tirn_h, tirn_l, Wk_h, Wk_l, bk, nullptr, Kp_h, Kp_l);

  // ---- pass 1: rgb_out = attn(q=rgb_n, kv, V=[rgb, tir]) @ Wo + bo ----
  gemm_nt_bias_kernel<false><<<dim3(BH * NQ / 128, DIM / 128), blk, 0, stream>>>(
      rgbn_h, rgbn_l, nullptr, nullptr, Wq_h, Wq_l, bq, nullptr, Q_h, Q_l);
  attn_kernel<<<dim3(NQ / 128, NHEADS, BH), blk, 0, stream>>>(
      Q_h, Q_l, Kp_h, Kp_l, attn_rgb, attn_tir, O_h, O_l);
  gemm_nt_bias_kernel<true><<<dim3(BH * NQ / 128, DIM / 128), blk, 0, stream>>>(
      O_h, O_l, nullptr, nullptr, Wo_h, Wo_l, bo, out, nullptr, nullptr);

  // ---- pass 2: tir_out uses the UPDATED rgb (rgb_out) as first V half ----
  gemm_nt_bias_kernel<false><<<dim3(BH * NQ / 128, DIM / 128), blk, 0, stream>>>(
      tirn_h, tirn_l, nullptr, nullptr, Wq_h, Wq_l, bq, nullptr, Q_h, Q_l);
  attn_kernel<<<dim3(NQ / 128, NHEADS, BH), blk, 0, stream>>>(
      Q_h, Q_l, Kp_h, Kp_l, out /* rgb_out */, attn_tir, O_h, O_l);
  gemm_nt_bias_kernel<true><<<dim3(BH * NQ / 128, DIM / 128), blk, 0, stream>>>(
      O_h, O_l, nullptr, nullptr, Wo_h, Wo_l, bo, out + HALF, nullptr, nullptr);
}
